// APPNPNet_90675349553255
// MI455X (gfx1250) — matrix operand order repaired and checked by reading, not compiled, not run on hardware
//
#include <hip/hip_runtime.h>
#include <hip/hip_bf16.h>

typedef __attribute__((ext_vector_type(16))) _Float16 v16h;
typedef __attribute__((ext_vector_type(8)))  _Float16 v8h;
typedef __attribute__((ext_vector_type(8)))  float    v8f;
typedef __attribute__((ext_vector_type(4)))  float    v4f;

#define IN_DIM  512
#define HID_DIM 256
#define OUT_DIM 64
#define ALPHA_F 0.1f
#define OMA_F   0.9f      // 1 - alpha
#define H1S     264       // padded LDS row stride (f16) to avoid bank conflicts

// ---------------------------------------------------------------------------
// Pack W1 [512,256] f32 row-major -> WMMA B-fragment-ordered f16.
// Fragment f = j*16 + kb  (j: n-block 0..15, kb: k-block 0..15), 512 f16 each.
// B layout (16-bit, 32x16): lane L -> column N = L%16; element e -> K = (e&7) + 8*(L>=16) + 16*(e>=8)
__global__ void pack_w1_k(const float* __restrict__ W1, _Float16* __restrict__ W1p) {
    int tid = blockIdx.x * 256 + threadIdx.x;
    if (tid >= 16 * 16 * 512) return;
    int f = tid >> 9, r = tid & 511;
    int lane = r >> 4, e = r & 15;
    int j = f >> 4, kb = f & 15;
    int n = j * 16 + (lane & 15);
    int k = kb * 32 + (e & 7) + ((lane >> 4) * 8) + ((e >> 3) * 16);
    W1p[tid] = (_Float16)W1[k * HID_DIM + n];
}

// Pack W2 [256,64] f32 -> fragments, f = j*8 + kb (j 0..3, kb 0..7)
__global__ void pack_w2_k(const float* __restrict__ W2, _Float16* __restrict__ W2p) {
    int tid = blockIdx.x * 256 + threadIdx.x;
    if (tid >= 4 * 8 * 512) return;
    int f = tid >> 9, r = tid & 511;
    int lane = r >> 4, e = r & 15;
    int j = f >> 3, kb = f & 7;
    int n = j * 16 + (lane & 15);
    int k = kb * 32 + (e & 7) + ((lane >> 4) * 8) + ((e >> 3) * 16);
    W2p[tid] = (_Float16)W2[k * OUT_DIM + n];
}

// ---------------------------------------------------------------------------
// Degree / norm
__global__ void deg_init_k(float* __restrict__ deg, long long N) {
    long long i = (long long)blockIdx.x * 256 + threadIdx.x;
    if (i < N) deg[i] = 1.0f;                    // self loop
}
__global__ void deg_count_k(const long long* __restrict__ dst, float* __restrict__ deg, long long E) {
    long long e = (long long)blockIdx.x * 256 + threadIdx.x;
    if (e < E) atomicAdd(deg + dst[e], 1.0f);
}
__global__ void dis_k(const float* __restrict__ deg, float* __restrict__ dis, long long N) {
    long long i = (long long)blockIdx.x * 256 + threadIdx.x;
    if (i < N) { float d = deg[i]; dis[i] = d > 0.0f ? rsqrtf(d) : 0.0f; }
}
__global__ void norm_k(const long long* __restrict__ src, const long long* __restrict__ dst,
                       const float* __restrict__ dis, float* __restrict__ norm, long long E) {
    long long e = (long long)blockIdx.x * 256 + threadIdx.x;
    if (e < E) norm[e] = dis[src[e]] * dis[dst[e]];
}

// ---------------------------------------------------------------------------
// Fused 2-layer MLP, one wave (32 threads) per 16-row tile. WMMA f32=f16*f16+f32.
__global__ void __launch_bounds__(32)
mlp_wmma_k(const float* __restrict__ x, const _Float16* __restrict__ W1p,
           const float* __restrict__ b1, const _Float16* __restrict__ W2p,
           const float* __restrict__ b2, float* __restrict__ h, int N) {
    __shared__ _Float16 h1[16 * H1S];            // 16 x 256 f16, padded stride

    const int tile = blockIdx.x;
    const int lane = threadIdx.x;                // 0..31 (wave32)
    const int mrow = lane & 15;                  // matrix row this lane feeds (A) / col (B,D)
    const int hiK  = (lane >> 4) * 8;            // A-layout K offset for lanes 16..31
    const int rbase = (lane >> 4) * 8;           // D-layout row offset for lanes 16..31

    int row = tile * 16 + mrow;
    int rr  = row < N ? row : N - 1;             // clamp loads on partial tile

    // ---- Layer 1: H1 = relu(x_tile @ W1 + b1), 16x256 ----
    for (int j = 0; j < 16; ++j) {               // output n-blocks
        v8f acc = {};
        for (int kb = 0; kb < 16; ++kb) {        // K blocks of 32
            // A fragment (16-bit A 16x32): lanes<16: K = 0..7,16..23; lanes>=16: +8
            const float* xp = x + (size_t)rr * IN_DIM + kb * 32 + hiK;
            v4f x0 = *(const v4f*)(xp);
            v4f x1 = *(const v4f*)(xp + 4);
            v4f x2 = *(const v4f*)(xp + 16);
            v4f x3 = *(const v4f*)(xp + 20);
            v16h a;
#pragma unroll
            for (int t = 0; t < 4; ++t) {
                a[t]      = (_Float16)x0[t];
                a[4 + t]  = (_Float16)x1[t];
                a[8 + t]  = (_Float16)x2[t];
                a[12 + t] = (_Float16)x3[t];
            }
            v16h b = *(const v16h*)(W1p + (size_t)((j * 16 + kb) * 32 + lane) * 16);
            acc = __builtin_amdgcn_wmma_f32_16x16x32_f16(
                false, a, false, b, (short)0, acc, false, false);
        }
        // D layout: VGPR r -> row rbase+r, col = mrow (within j-block)
        int col = j * 16 + mrow;
        float bias = b1[col];
#pragma unroll
        for (int r = 0; r < 8; ++r) {
            float v = acc[r] + bias;
            h1[(rbase + r) * H1S + col] = (_Float16)(v > 0.0f ? v : 0.0f);
        }
    }
    __syncthreads();

    // ---- Layer 2: out = H1 @ W2 + b2, 16x64 ----
    for (int j = 0; j < 4; ++j) {
        v8f acc = {};
        for (int kb = 0; kb < 8; ++kb) {
            const _Float16* hp = h1 + mrow * H1S + kb * 32 + hiK;
            v8h a0 = *(const v8h*)(hp);          // K = base..base+7
            v8h a1 = *(const v8h*)(hp + 16);     // K = base+16..base+23
            v16h a;
#pragma unroll
            for (int t = 0; t < 8; ++t) { a[t] = a0[t]; a[8 + t] = a1[t]; }
            v16h b = *(const v16h*)(W2p + (size_t)((j * 8 + kb) * 32 + lane) * 16);
            acc = __builtin_amdgcn_wmma_f32_16x16x32_f16(
                false, a, false, b, (short)0, acc, false, false);
        }
        int col = j * 16 + mrow;
        float bias = b2[col];
#pragma unroll
        for (int r = 0; r < 8; ++r) {
            int row_o = tile * 16 + rbase + r;
            if (row_o < N) h[(size_t)row_o * OUT_DIM + col] = acc[r] + bias;
        }
    }
}

// ---------------------------------------------------------------------------
// Propagation step: zout = alpha*h + (1-alpha)*(self-loop term), then edge scatter
__global__ void prop_init_k(const float* __restrict__ h, const float* __restrict__ zin,
                            const float* __restrict__ dis, float* __restrict__ zout,
                            long long total) {
    long long i = (long long)blockIdx.x * 256 + threadIdx.x;
    if (i >= total) return;
    long long row = i >> 6;
    float di = dis[row];
    zout[i] = ALPHA_F * h[i] + OMA_F * di * di * zin[i];
}

// One wave covers 32 contiguous channels of one edge -> 128B coalesced z-row gather,
// z (25.6MB) is L2-resident so the 10x845MB gather/scatter stays on-chip.
__global__ void prop_scatter_k(const long long* __restrict__ src, const long long* __restrict__ dst,
                               const float* __restrict__ norm, const float* __restrict__ zin,
                               float* __restrict__ zout, long long E) {
    long long idx = (long long)blockIdx.x * 256 + threadIdx.x;
    long long e = idx >> 6;
    if (e >= E) return;
    int c = (int)(idx & 63);
    long long s = src[e], d = dst[e];
    float w = OMA_F * norm[e];
    atomicAdd(zout + d * OUT_DIM + c, w * zin[s * OUT_DIM + c]);
}

// ---------------------------------------------------------------------------
// log_softmax over 64 channels: one wave per row, lane owns c and c+32
__global__ void logsoftmax_k(float* __restrict__ z, int N) {
    int row = blockIdx.x * 8 + (threadIdx.x >> 5);
    if (row >= N) return;
    int lane = threadIdx.x & 31;
    float* p = z + (size_t)row * OUT_DIM;
    float a0 = p[lane], a1 = p[lane + 32];
    float m = fmaxf(a0, a1);
#pragma unroll
    for (int o = 16; o > 0; o >>= 1) m = fmaxf(m, __shfl_xor(m, o, 32));
    float s = __expf(a0 - m) + __expf(a1 - m);
#pragma unroll
    for (int o = 16; o > 0; o >>= 1) s += __shfl_xor(s, o, 32);
    float ls = m + __logf(s);
    p[lane]      = a0 - ls;
    p[lane + 32] = a1 - ls;
}

// ---------------------------------------------------------------------------
extern "C" void kernel_launch(void* const* d_in, const int* in_sizes, int n_in,
                              void* d_out, int out_size, void* d_ws, size_t ws_size,
                              hipStream_t stream) {
    const float*     x  = (const float*)d_in[0];
    const long long* ei = (const long long*)d_in[1];   // int64 per reference
    const float*     W1 = (const float*)d_in[2];
    const float*     b1 = (const float*)d_in[3];
    const float*     W2 = (const float*)d_in[4];
    const float*     b2 = (const float*)d_in[5];
    float*           out = (float*)d_out;

    const long long N = in_sizes[0] / IN_DIM;
    const long long E = in_sizes[1] / 2;
    const long long* srcp = ei;
    const long long* dstp = ei + E;

    // carve workspace (256B aligned regions)
    char* ws = (char*)d_ws;
    size_t off = 0;
    auto carve = [&](size_t bytes) -> void* {
        void* p = ws + off;
        off += (bytes + 255) & ~(size_t)255;
        return p;
    };
    _Float16* W1p = (_Float16*)carve(sizeof(_Float16) * IN_DIM * HID_DIM);
    _Float16* W2p = (_Float16*)carve(sizeof(_Float16) * HID_DIM * OUT_DIM);
    float* deg  = (float*)carve(sizeof(float) * N);
    float* dis  = (float*)carve(sizeof(float) * N);
    float* norm = (float*)carve(sizeof(float) * E);
    float* h    = (float*)carve(sizeof(float) * N * OUT_DIM);
    float* z1   = (float*)carve(sizeof(float) * N * OUT_DIM);
    (void)ws_size; (void)n_in; (void)out_size;

    // 1) pack weights to WMMA fragment layout (f16)
    pack_w1_k<<<(16 * 16 * 512 + 255) / 256, 256, 0, stream>>>(W1, W1p);
    pack_w2_k<<<(4 * 8 * 512 + 255) / 256, 256, 0, stream>>>(W2, W2p);

    // 2) gcn_norm
    unsigned gN = (unsigned)((N + 255) / 256);
    unsigned gE = (unsigned)((E + 255) / 256);
    deg_init_k <<<gN, 256, 0, stream>>>(deg, N);
    deg_count_k<<<gE, 256, 0, stream>>>(dstp, deg, E);
    dis_k      <<<gN, 256, 0, stream>>>(deg, dis, N);
    norm_k     <<<gE, 256, 0, stream>>>(srcp, dstp, dis, norm, E);

    // 3) fused MLP via WMMA: one wave per 16-row tile
    mlp_wmma_k<<<(unsigned)((N + 15) / 16), 32, 0, stream>>>(x, W1p, b1, W2p, b2, h, (int)N);

    // 4) K=10 PPR steps, ping-pong between z1 and d_out (step 9 ends in d_out)
    const long long totalNC = N * OUT_DIM;
    const long long totalEC = E * OUT_DIM;
    unsigned gNC = (unsigned)((totalNC + 255) / 256);
    unsigned gEC = (unsigned)((totalEC + 255) / 256);
    float* zbuf[2] = {z1, out};
    for (int s = 0; s < 10; ++s) {
        const float* zin = (s == 0) ? h : zbuf[(s + 1) & 1];
        float* zout = zbuf[s & 1];
        prop_init_k   <<<gNC, 256, 0, stream>>>(h, zin, dis, zout, totalNC);
        prop_scatter_k<<<gEC, 256, 0, stream>>>(srcp, dstp, norm, zin, zout, E);
    }

    // 5) log_softmax in place on d_out
    logsoftmax_k<<<(unsigned)((N + 7) / 8), 256, 0, stream>>>(out, (int)N);
}
